// KeypointPipeline_63608465654139
// MI455X (gfx1250) — matrix operand order repaired and checked by reading, not compiled, not run on hardware
//
#include <hip/hip_runtime.h>

// ---------------------------------------------------------------------------
// Fused NRI encoder + decoder step for gfx1250 (MI455X), wave32 + WMMA f16.
// Weights are pre-transposed to f16 WMMA images in d_ws by tiny prep kernels
// (done once per launch, trivially cheap); the fused kernel then stages each
// stage's weight blob into a double-buffered LDS slot with software-pipelined
// copies (global_load_b128 issued before the current stage's GEMM, committed
// to LDS after), overlapping weight movement with WMMA compute.
// ---------------------------------------------------------------------------

typedef __attribute__((ext_vector_type(16))) _Float16 v16h;
typedef __attribute__((ext_vector_type(8)))  _Float16 v8h;
typedef __attribute__((ext_vector_type(8)))  float    v8f;

#define BDIM     384          // 12 wave32
#define BT       32           // graphs per workgroup
#define NNODE    6
#define ROWS     (BT * NNODE) // 192 rows
#define HDIM     128
#define DIN      3

// ---- LDS arena (bytes) ----
#define OFF_A    0            // act buffer A: 192 x 128 f16   (49152)
#define OFF_B    49152        // act buffer B: 192 x 128 f16   (49152)
#define OFF_W0   98304        // weight slot 0 (<= 64KB)
#define OFF_W1   163840       // weight slot 1 (<= 64KB)
#define OFF_WS   229376       // small persistent weights (mw1 pads, 16KB)
#define OFF_X    245760       // node x rows, K=32 padded: 192 x 32 f16
#define OFF_XE   258048       // edge [x_recv|x_send] rows: 192 x 32 f16
#define OFF_RT   270336       // rel_type / logits: 192 x 2 f32
#define OFF_TMP  271872       // per-wave msg scratch: 12 x (16 x 128) f16
#define LDS_BYTES 321024

// ---- d_ws weight-image layout (in f16 halves) ----
#define WS_B0   0        //  4096  e1w1  [128][32]  K-pad
#define WS_B1   4096     // 16384  e1w2  [128][128]
#define WS_B2   20480    // 32768  e2w1  a|b
#define WS_B3   53248    // 16384  e2w2
#define WS_B4   69632    // 16384  e3w1
#define WS_B5   86016    // 16384  e3w2
#define WS_B6   102400   // 32768  e4w1  a|b
#define WS_B7   135168   // 16384  e4w2
#define WS_B8   151552   //  2048  ew_out [16][128] N-pad
#define WS_B9A  153600   //  8192  mw1 t0|t1 [128][32] K-pad
#define WS_B9B  161792   // 16384  mw2 t0
#define WS_B10  178176   // 16384  mw2 t1
#define WS_B11  194560   // 20480  ow1x pad | ow1a
#define WS_B12  215040   // 16384  ow2
#define WS_B13  231424   //  2048  ow3 [16][128] N-pad
#define WS_TOTAL_HALVES 233472

struct Params {
  const float *x;
  const float *e1b1,*e1b2,*e2b1,*e2b2,*e3b1,*e3b2,*e4b1,*e4b2;
  const float *eb_out,*mb1,*mb2,*ob1,*ob2,*ob3;
  const _Float16 *ws;          // pre-transposed f16 weight images
  float *out;
  int nBatch;
};

__device__ __forceinline__ v8f vzero8() {
  v8f z = {0.f,0.f,0.f,0.f,0.f,0.f,0.f,0.f};
  return z;
}

// ---- software-pipelined weight copy (global -> regs -> LDS) ----
template<int HALVES>
struct PF {
  static constexpr int NVEC = HALVES / 8;
  static constexpr int NV   = (NVEC + BDIM - 1) / BDIM;
  v8h v[NV];
  __device__ __forceinline__ void load(const _Float16* g) {
#pragma unroll
    for (int j = 0; j < NV; ++j) {
      int i = (int)threadIdx.x + j * BDIM;
      if (i < NVEC) v[j] = ((const v8h*)g)[i];
    }
  }
  __device__ __forceinline__ void store(_Float16* l) const {
#pragma unroll
    for (int j = 0; j < NV; ++j) {
      int i = (int)threadIdx.x + j * BDIM;
      if (i < NVEC) ((v8h*)l)[i] = v[j];
    }
  }
};

// A fragment: lane l holds row M=l&15; halves K = kbase+sel16+{0..15}
// (sel16 = (l>=16)?16:0).
__device__ __forceinline__ v16h ld_a(const _Float16* p) {
  v8h lo = *(const v8h*)(p);
  v8h hi = *(const v8h*)(p + 8);
  v16h a;
#pragma unroll
  for (int j = 0; j < 8; ++j) { a[j] = lo[j]; a[8 + j] = hi[j]; }
  return a;
}

// B fragment from transposed weights WT[N][K].
__device__ __forceinline__ v16h ld_b(const _Float16* p) {
  v8h lo = *(const v8h*)(p);
  v8h hi = *(const v8h*)(p + 8);
  v16h b;
#pragma unroll
  for (int j = 0; j < 8; ++j) { b[j] = lo[j]; b[8 + j] = hi[j]; }
  return b;
}

template<int KT, int NT>
__device__ __forceinline__ void gemm_acc(v8f* acc, const _Float16* aRowLane,
                                         const _Float16* wT, int wStride, int lane) {
  const int sel8  = (lane & 16) ? 8  : 0;
  const int sel16 = (lane & 16) ? 16 : 0;
  const int n0    = lane & 15;
#pragma unroll
  for (int kt = 0; kt < KT; ++kt) {
    v16h a = ld_a(aRowLane + kt * 32 + sel16);
#pragma unroll
    for (int nt = 0; nt < NT; ++nt) {
      v16h b = ld_b(wT + (nt * 16 + n0) * wStride + kt * 32 + sel16);
      acc[nt] = __builtin_amdgcn_wmma_f32_16x16x32_f16(
          false, a, false, b, (short)0, acc[nt], false, false);
    }
  }
}

// MODE: 0 = identity, 1 = relu, 2 = elu
template<int MODE>
__device__ __forceinline__ void epilogue_store(const v8f* acc, const float* bias,
                                               _Float16* dst, int row0, int stride,
                                               int lane) {
  const int selR = (lane & 16) ? 8 : 0;
  const int n0   = lane & 15;
#pragma unroll
  for (int nt = 0; nt < 8; ++nt) {
    float b = bias[nt * 16 + n0];
#pragma unroll
    for (int i = 0; i < 8; ++i) {
      float v = acc[nt][i] + b;
      if (MODE == 1) v = fmaxf(v, 0.f);
      if (MODE == 2) v = (v > 0.f) ? v : (__expf(v) - 1.f);
      dst[(row0 + i + selR) * stride + nt * 16 + n0] = (_Float16)v;
    }
  }
}

// ---- prep kernels: build transposed f16 weight images in d_ws ----
extern "C" __global__ void prep_wT(const float* __restrict__ W,
                                   _Float16* __restrict__ dst, int Kreal, int Kp) {
  int total = HDIM * Kp;
  for (int idx = blockIdx.x * blockDim.x + threadIdx.x; idx < total;
       idx += gridDim.x * blockDim.x) {
    int n = idx / Kp, k = idx - n * Kp;
    dst[idx] = (k < Kreal) ? (_Float16)W[k * HDIM + n] : (_Float16)0.f;
  }
}

extern "C" __global__ void prep_wN(const float* __restrict__ W,
                                   _Float16* __restrict__ dst, int Nreal) {
  int total = 16 * HDIM;
  for (int idx = blockIdx.x * blockDim.x + threadIdx.x; idx < total;
       idx += gridDim.x * blockDim.x) {
    int n = idx / HDIM, k = idx - n * HDIM;
    dst[idx] = (n < Nreal) ? (_Float16)W[k * Nreal + n] : (_Float16)0.f;
  }
}

extern "C" __global__ __launch_bounds__(BDIM) void nri_fused(Params p) {
  extern __shared__ char smem[];
  _Float16* bufA = (_Float16*)(smem + OFF_A);
  _Float16* bufB = (_Float16*)(smem + OFF_B);
  _Float16* sW0  = (_Float16*)(smem + OFF_W0);
  _Float16* sW1  = (_Float16*)(smem + OFF_W1);
  _Float16* sWs  = (_Float16*)(smem + OFF_WS);
  _Float16* sX   = (_Float16*)(smem + OFF_X);
  _Float16* sXE  = (_Float16*)(smem + OFF_XE);
  float*    sRT  = (float*)   (smem + OFF_RT);

  const int tid  = threadIdx.x;
  const int lane = tid & 31;
  const int wave = tid >> 5;          // 0..11 == M-tile index
  const int t    = wave;
  _Float16* sTmp = (_Float16*)(smem + OFF_TMP) + wave * (16 * HDIM);

  const int wgB0 = blockIdx.x * BT;
  if (wgB0 >= p.nBatch) return;
  const float* gx = p.x + (size_t)wgB0 * NNODE * DIN;
  const _Float16* ws = p.ws;
  const int n0   = lane & 15;
  const int selR = (lane & 16) ? 8 : 0;

  // ---- S0: stage x; sync-load blob0 into slot0 ----
  { PF<4096> pf; pf.load(ws + WS_B0); pf.store(sW0); }
  for (int idx = tid; idx < ROWS * 32; idx += BDIM) {
    int r = idx >> 5, c = idx & 31;
    sX[idx] = (c < DIN) ? (_Float16)gx[r * DIN + c] : (_Float16)0.f;
    int b = r / NNODE, e = r - b * NNODE;
    int rr = b * NNODE + (e + 1) % NNODE;        // recv node of edge e
    float ve = 0.f;
    if (c < DIN)           ve = gx[rr * DIN + c];
    else if (c < 2 * DIN)  ve = gx[r * DIN + (c - DIN)];
    sXE[idx] = (_Float16)ve;
  }
  __syncthreads();

  // ---- S1: h' = elu(x @ e1w1 + b) -> bufA   [W: slot0, pf e1w2 -> slot1] --
  {
    PF<16384> pf; pf.load(ws + WS_B1);
    v8f acc[8];
#pragma unroll
    for (int q = 0; q < 8; ++q) acc[q] = vzero8();
    gemm_acc<1, 8>(acc, sX + (t * 16 + n0) * 32, sW0, 32, lane);
    epilogue_store<2>(acc, p.e1b1, bufA, t * 16, HDIM, lane);
    pf.store(sW1);
  }
  __syncthreads();

  // ---- S2: h = elu(h' @ e1w2 + b) -> bufB   [W: slot1, pf e2w1 -> slot0] --
  {
    PF<32768> pf; pf.load(ws + WS_B2);
    v8f acc[8];
#pragma unroll
    for (int q = 0; q < 8; ++q) acc[q] = vzero8();
    gemm_acc<4, 8>(acc, bufA + (t * 16 + n0) * HDIM, sW1, HDIM, lane);
    epilogue_store<2>(acc, p.e1b2, bufB, t * 16, HDIM, lane);
    pf.store(sW0);
  }
  __syncthreads();

  // ---- S3: elu(h_recv@Wa + h_send@Wb + b) -> bufA  [slot0, pf -> slot1] ---
  {
    PF<16384> pf; pf.load(ws + WS_B3);
    v8f acc[8];
#pragma unroll
    for (int q = 0; q < 8; ++q) acc[q] = vzero8();
    int r = t * 16 + n0, b = r / NNODE, e = r - b * NNODE;
    gemm_acc<4, 8>(acc, bufB + (b * NNODE + (e + 1) % NNODE) * HDIM, sW0, HDIM, lane);
    gemm_acc<4, 8>(acc, bufB + r * HDIM, sW0 + 16384, HDIM, lane);
    epilogue_store<2>(acc, p.e2b1, bufA, t * 16, HDIM, lane);
    pf.store(sW1);
  }
  __syncthreads();

  // ---- S4: he = elu(. @ e2w2 + b) -> bufB   [slot1, pf e3w1 -> slot0] ----
  {
    PF<16384> pf; pf.load(ws + WS_B4);
    v8f acc[8];
#pragma unroll
    for (int q = 0; q < 8; ++q) acc[q] = vzero8();
    gemm_acc<4, 8>(acc, bufA + (t * 16 + n0) * HDIM, sW1, HDIM, lane);
    epilogue_store<2>(acc, p.e2b2, bufB, t * 16, HDIM, lane);
    pf.store(sW0);
  }
  __syncthreads();

  // ---- S5: elu(he @ e3w1 + b) -> bufA       [slot0, pf e3w2 -> slot1] ----
  {
    PF<16384> pf; pf.load(ws + WS_B5);
    v8f acc[8];
#pragma unroll
    for (int q = 0; q < 8; ++q) acc[q] = vzero8();
    gemm_acc<4, 8>(acc, bufB + (t * 16 + n0) * HDIM, sW0, HDIM, lane);
    epilogue_store<2>(acc, p.e3b1, bufA, t * 16, HDIM, lane);
    pf.store(sW1);
  }
  __syncthreads();

  // ---- S6: h2 = elu(. @ e3w2 + b) -> bufA in-place [slot1, pf -> slot0] --
  {
    PF<32768> pf; pf.load(ws + WS_B6);
    v8f acc[8];
#pragma unroll
    for (int q = 0; q < 8; ++q) acc[q] = vzero8();
    gemm_acc<4, 8>(acc, bufA + (t * 16 + n0) * HDIM, sW1, HDIM, lane);
    epilogue_store<2>(acc, p.e3b2, bufA, t * 16, HDIM, lane);
    pf.store(sW0);
  }
  __syncthreads();

  // ---- S7: elu(h2@Wa + he@Wb + b) -> bufB in-place [slot0, pf -> slot1] --
  {
    PF<16384> pf; pf.load(ws + WS_B7);
    v8f acc[8];
#pragma unroll
    for (int q = 0; q < 8; ++q) acc[q] = vzero8();
    gemm_acc<4, 8>(acc, bufA + (t * 16 + n0) * HDIM, sW0, HDIM, lane);
    gemm_acc<4, 8>(acc, bufB + (t * 16 + n0) * HDIM, sW0 + 16384, HDIM, lane);
    epilogue_store<2>(acc, p.e4b1, bufB, t * 16, HDIM, lane);
    pf.store(sW1);
  }
  __syncthreads();

  // ---- S8: h3 = elu(. @ e4w2 + b) -> bufA   [slot1, pf ew_out -> slot0] --
  {
    PF<2048> pf; pf.load(ws + WS_B8);
    v8f acc[8];
#pragma unroll
    for (int q = 0; q < 8; ++q) acc[q] = vzero8();
    gemm_acc<4, 8>(acc, bufB + (t * 16 + n0) * HDIM, sW1, HDIM, lane);
    epilogue_store<2>(acc, p.e4b2, bufA, t * 16, HDIM, lane);
    pf.store(sW0);
  }
  __syncthreads();

  // ---- S9: logits via WMMA (N pad 2->16) -> sRT
  //          [slot0; pf mw1 pads -> sWs, mw2 t0 -> slot1] ----
  {
    PF<8192>  pfa; pfa.load(ws + WS_B9A);
    PF<16384> pfb; pfb.load(ws + WS_B9B);
    v8f acc[1] = { vzero8() };
    gemm_acc<4, 1>(acc, bufA + (t * 16 + n0) * HDIM, sW0, HDIM, lane);
    if (n0 < 2) {
      float b = p.eb_out[n0];
#pragma unroll
      for (int i = 0; i < 8; ++i)
        sRT[(t * 16 + i + selR) * 2 + n0] = acc[0][i] + b;
    }
    pfa.store(sWs);
    pfb.store(sW1);
  }
  __syncthreads();
  for (int r = tid; r < ROWS; r += BDIM) {
    float l0 = sRT[r * 2 + 0], l1 = sRT[r * 2 + 1];
    float m  = fmaxf(l0, l1);
    float e0 = __expf(l0 - m), e1 = __expf(l1 - m);
    float s  = e0 + e1;
    sRT[r * 2 + 0] = e0 / s;
    sRT[r * 2 + 1] = e1 / s;
  }
  __syncthreads();

  // ---- S10: per-type msg MLPs, rel_type-weighted accumulate -> bufB ----
  v8f msg[8];
#pragma unroll
  for (int q = 0; q < 8; ++q) msg[q] = vzero8();
  {
    // type 0: mw1 t0 in sWs, mw2 t0 in slot1; pf mw2 t1 -> slot0
    PF<16384> pf; pf.load(ws + WS_B10);
    v8f a1[8];
#pragma unroll
    for (int q = 0; q < 8; ++q) a1[q] = vzero8();
    gemm_acc<1, 8>(a1, sXE + (t * 16 + n0) * 32, sWs, 32, lane);
    epilogue_store<1>(a1, p.mb1, sTmp, 0, HDIM, lane);
    v8f a2[8];
#pragma unroll
    for (int q = 0; q < 8; ++q) a2[q] = vzero8();
    gemm_acc<4, 8>(a2, sTmp + n0 * HDIM, sW1, HDIM, lane);
    float rt[8];
#pragma unroll
    for (int i = 0; i < 8; ++i) rt[i] = sRT[(t * 16 + i + selR) * 2 + 0];
#pragma unroll
    for (int nt = 0; nt < 8; ++nt) {
      float b = p.mb2[nt * 16 + n0];
#pragma unroll
      for (int i = 0; i < 8; ++i)
        msg[nt][i] += rt[i] * fmaxf(a2[nt][i] + b, 0.f);
    }
    pf.store(sW0);
  }
  __syncthreads();
  {
    // type 1: mw1 t1 in sWs+4096, mw2 t1 in slot0; pf ow1 -> slot1
    PF<20480> pf; pf.load(ws + WS_B11);
    v8f a1[8];
#pragma unroll
    for (int q = 0; q < 8; ++q) a1[q] = vzero8();
    gemm_acc<1, 8>(a1, sXE + (t * 16 + n0) * 32, sWs + 4096, 32, lane);
    epilogue_store<1>(a1, p.mb1 + HDIM, sTmp, 0, HDIM, lane);
    v8f a2[8];
#pragma unroll
    for (int q = 0; q < 8; ++q) a2[q] = vzero8();
    gemm_acc<4, 8>(a2, sTmp + n0 * HDIM, sW0, HDIM, lane);
    float rt[8];
#pragma unroll
    for (int i = 0; i < 8; ++i) rt[i] = sRT[(t * 16 + i + selR) * 2 + 1];
#pragma unroll
    for (int nt = 0; nt < 8; ++nt) {
      float b = p.mb2[HDIM + nt * 16 + n0];
#pragma unroll
      for (int i = 0; i < 8; ++i)
        msg[nt][i] += rt[i] * fmaxf(a2[nt][i] + b, 0.f);
    }
#pragma unroll
    for (int nt = 0; nt < 8; ++nt)
#pragma unroll
      for (int i = 0; i < 8; ++i)
        bufB[(t * 16 + i + selR) * HDIM + nt * 16 + n0] = (_Float16)msg[nt][i];
    pf.store(sW1);
  }
  __syncthreads();

  // ---- S11: relu([x|agg] @ d_out_w1 + b) -> bufA; agg[n] = msg[(n+5)%6]
  //           [slot1: ow1x @ +0 (stride 32), ow1a @ +4096; pf ow2 -> slot0] -
  {
    PF<16384> pf; pf.load(ws + WS_B12);
    v8f acc[8];
#pragma unroll
    for (int q = 0; q < 8; ++q) acc[q] = vzero8();
    int r = t * 16 + n0, b = r / NNODE, n = r - b * NNODE;
    gemm_acc<1, 8>(acc, sX + r * 32, sW1, 32, lane);
    gemm_acc<4, 8>(acc, bufB + (b * NNODE + (n + 5) % NNODE) * HDIM,
                   sW1 + 4096, HDIM, lane);
    epilogue_store<1>(acc, p.ob1, bufA, t * 16, HDIM, lane);
    pf.store(sW0);
  }
  __syncthreads();

  // ---- S12: relu(. @ d_out_w2 + b) -> bufB  [slot0, pf ow3 -> slot1] ----
  {
    PF<2048> pf; pf.load(ws + WS_B13);
    v8f acc[8];
#pragma unroll
    for (int q = 0; q < 8; ++q) acc[q] = vzero8();
    gemm_acc<4, 8>(acc, bufA + (t * 16 + n0) * HDIM, sW0, HDIM, lane);
    epilogue_store<1>(acc, p.ob2, bufB, t * 16, HDIM, lane);
    pf.store(sW1);
  }
  __syncthreads();

  // ---- S13: out = x + (. @ d_out_w3 + b3) via WMMA (N pad 3->16) [slot1] --
  {
    v8f acc[1] = { vzero8() };
    gemm_acc<4, 1>(acc, bufB + (t * 16 + n0) * HDIM, sW1, HDIM, lane);
    if (n0 < DIN) {
      float b = p.ob3[n0];
#pragma unroll
      for (int i = 0; i < 8; ++i) {
        int r = t * 16 + i + selR;
        float v = acc[0][i] + b + gx[r * DIN + n0];
        p.out[((size_t)wgB0 * NNODE + r) * DIN + n0] = v;
      }
    }
  }
}

extern "C" void kernel_launch(void* const* d_in, const int* in_sizes, int n_in,
                              void* d_out, int out_size, void* d_ws, size_t ws_size,
                              hipStream_t stream) {
  (void)n_in; (void)out_size; (void)ws_size;
  const float* x    = (const float*)d_in[0];
  const float* e1w1 = (const float*)d_in[3],  *e1b1 = (const float*)d_in[4];
  const float* e1w2 = (const float*)d_in[5],  *e1b2 = (const float*)d_in[6];
  const float* e2w1 = (const float*)d_in[7],  *e2b1 = (const float*)d_in[8];
  const float* e2w2 = (const float*)d_in[9],  *e2b2 = (const float*)d_in[10];
  const float* e3w1 = (const float*)d_in[11], *e3b1 = (const float*)d_in[12];
  const float* e3w2 = (const float*)d_in[13], *e3b2 = (const float*)d_in[14];
  const float* e4w1 = (const float*)d_in[15], *e4b1 = (const float*)d_in[16];
  const float* e4w2 = (const float*)d_in[17], *e4b2 = (const float*)d_in[18];
  const float* ew_out = (const float*)d_in[19], *eb_out = (const float*)d_in[20];
  const float* mw1  = (const float*)d_in[21], *mb1 = (const float*)d_in[22];
  const float* mw2  = (const float*)d_in[23], *mb2 = (const float*)d_in[24];
  const float* ow1  = (const float*)d_in[25], *ob1 = (const float*)d_in[26];
  const float* ow2  = (const float*)d_in[27], *ob2 = (const float*)d_in[28];
  const float* ow3  = (const float*)d_in[29], *ob3 = (const float*)d_in[30];
  _Float16* ws = (_Float16*)d_ws;

  // ---- build transposed f16 weight images in d_ws ----
  dim3 pg(64), pb(256);
  prep_wT<<<pg, pb, 0, stream>>>(e1w1, ws + WS_B0, DIN, 32);
  prep_wT<<<pg, pb, 0, stream>>>(e1w2, ws + WS_B1, HDIM, HDIM);
  prep_wT<<<pg, pb, 0, stream>>>(e2w1,              ws + WS_B2,         HDIM, HDIM);
  prep_wT<<<pg, pb, 0, stream>>>(e2w1 + HDIM*HDIM,  ws + WS_B2 + 16384, HDIM, HDIM);
  prep_wT<<<pg, pb, 0, stream>>>(e2w2, ws + WS_B3, HDIM, HDIM);
  prep_wT<<<pg, pb, 0, stream>>>(e3w1, ws + WS_B4, HDIM, HDIM);
  prep_wT<<<pg, pb, 0, stream>>>(e3w2, ws + WS_B5, HDIM, HDIM);
  prep_wT<<<pg, pb, 0, stream>>>(e4w1,              ws + WS_B6,         HDIM, HDIM);
  prep_wT<<<pg, pb, 0, stream>>>(e4w1 + HDIM*HDIM,  ws + WS_B6 + 16384, HDIM, HDIM);
  prep_wT<<<pg, pb, 0, stream>>>(e4w2, ws + WS_B7, HDIM, HDIM);
  prep_wN<<<pg, pb, 0, stream>>>(ew_out, ws + WS_B8, 2);
  prep_wT<<<pg, pb, 0, stream>>>(mw1,            ws + WS_B9A,        2*DIN, 32);
  prep_wT<<<pg, pb, 0, stream>>>(mw1 + 6*HDIM,   ws + WS_B9A + 4096, 2*DIN, 32);
  prep_wT<<<pg, pb, 0, stream>>>(mw2,              ws + WS_B9B, HDIM, HDIM);
  prep_wT<<<pg, pb, 0, stream>>>(mw2 + HDIM*HDIM,  ws + WS_B10, HDIM, HDIM);
  prep_wT<<<pg, pb, 0, stream>>>(ow1,            ws + WS_B11,        DIN, 32);
  prep_wT<<<pg, pb, 0, stream>>>(ow1 + DIN*HDIM, ws + WS_B11 + 4096, HDIM, HDIM);
  prep_wT<<<pg, pb, 0, stream>>>(ow2, ws + WS_B12, HDIM, HDIM);
  prep_wN<<<pg, pb, 0, stream>>>(ow3, ws + WS_B13, DIN);

  Params p;
  p.x = x;
  p.e1b1 = e1b1; p.e1b2 = e1b2; p.e2b1 = e2b1; p.e2b2 = e2b2;
  p.e3b1 = e3b1; p.e3b2 = e3b2; p.e4b1 = e4b1; p.e4b2 = e4b2;
  p.eb_out = eb_out; p.mb1 = mb1; p.mb2 = mb2;
  p.ob1 = ob1; p.ob2 = ob2; p.ob3 = ob3;
  p.ws = ws;
  p.out = (float*)d_out;
  p.nBatch = in_sizes[0] / (NNODE * DIN);

  hipFuncSetAttribute((const void*)nri_fused,
                      hipFuncAttributeMaxDynamicSharedMemorySize, LDS_BYTES);
  int grid = (p.nBatch + BT - 1) / BT;
  nri_fused<<<grid, BDIM, LDS_BYTES, stream>>>(p);
}
